// SwinTransformerBlock_10075993277133
// MI455X (gfx1250) — compile-verified, matrix operand order checked by reading, not run on hardware
//
#include <hip/hip_runtime.h>
#include <hip/hip_bf16.h>
#include <math.h>

// ---------------------------------------------------------------------------
// Types / WMMA
// ---------------------------------------------------------------------------
typedef _Float16 half16 __attribute__((ext_vector_type(16)));
typedef _Float16 half8  __attribute__((ext_vector_type(8)));
typedef float    float8 __attribute__((ext_vector_type(8)));

static __device__ __forceinline__ float8 wmma_f16(half16 a, half16 b, float8 c) {
    return __builtin_amdgcn_wmma_f32_16x16x32_f16(false, a, false, b, (short)0, c, false, false);
}

// Build the 16-half A/B operand from LDS: elements 0..7 = K[base+0..7],
// elements 8..15 = K[base+16..23] (lanes>=16 use base+8, giving K 8..15 / 24..31).
static __device__ __forceinline__ half16 load_op(const _Float16* p) {
    half8 lo = *(const half8*)p;
    half8 hi = *(const half8*)(p + 16);
    half16 r;
#pragma unroll
    for (int i = 0; i < 8; ++i) { r[i] = lo[i]; r[i + 8] = hi[i]; }
    return r;
}

// LDS byte offset of a __shared__ object (generic ptr low 32 bits = LDS addr)
static __device__ __forceinline__ unsigned lds_off(const void* p) {
    return (unsigned)(unsigned long long)p;
}

// ---------------------------------------------------------------------------
// Problem constants
// ---------------------------------------------------------------------------
#define Bsz   32
#define Hh    56
#define Ww    56
#define Cc    256
#define NHh   8
#define HDd   32
#define WSw   7
#define Nn    49              // WS*WS
#define TW    2048            // total windows = B * 8 * 8
#define MR    100352          // TW * N  (GEMM row count)

// window-order row -> x-space row (shared by forward shift and reverse roll:
// spatial = (shifted + SS) mod 56 in both directions)
static __device__ __forceinline__ int win_to_x_row(int r) {
    int w   = r / Nn;
    int pos = r - w * Nn;
    int b   = w >> 6;
    int wy  = (w >> 3) & 7;
    int wx  = w & 7;
    int py  = pos / WSw;
    int px  = pos - py * WSw;
    int hs  = wy * WSw + py;
    int vs  = wx * WSw + px;
    int h   = hs + 3; if (h >= Hh) h -= Hh;
    int v   = vs + 3; if (v >= Ww) v -= Ww;
    return (b * Hh + h) * Ww + v;
}

// ---------------------------------------------------------------------------
// Weight transpose + fp32 -> f16  (Wt[n*K+k] = W[k*N+n])
// ---------------------------------------------------------------------------
__global__ void wt_convert(const float* __restrict__ W, _Float16* __restrict__ Wt,
                           int K, int N) {
    int idx = blockIdx.x * 256 + threadIdx.x;
    if (idx < K * N) {
        int k = idx / N, n = idx - k * N;
        Wt[(size_t)n * K + k] = (_Float16)W[idx];
    }
}

// ---------------------------------------------------------------------------
// LayerNorm (one block = one row of 256), optional window-gather mapping.
// mapmode 1: out row r (window order) <- LN(in[win_to_x_row(r)])  (LN1 + shift + partition)
// mapmode 0: out row r <- LN(in[r])                                (LN2)
// ---------------------------------------------------------------------------
static __device__ __forceinline__ float block_sum_256(float v, float* sbuf) {
#pragma unroll
    for (int o = 16; o > 0; o >>= 1) v += __shfl_down(v, o, 32);
    if ((threadIdx.x & 31) == 0) sbuf[threadIdx.x >> 5] = v;
    __syncthreads();
    float r = (threadIdx.x < 8) ? sbuf[threadIdx.x] : 0.0f;
#pragma unroll
    for (int o = 4; o > 0; o >>= 1) r += __shfl_down(r, o, 32);
    if (threadIdx.x == 0) sbuf[0] = r;
    __syncthreads();
    float res = sbuf[0];
    __syncthreads();
    return res;
}

__global__ __launch_bounds__(256) void ln_kernel(const float* __restrict__ in,
                                                 const float* __restrict__ g,
                                                 const float* __restrict__ b,
                                                 _Float16* __restrict__ out,
                                                 int mapmode) {
    __shared__ float sbuf[8];
    int r   = blockIdx.x;
    int src = (mapmode == 1) ? win_to_x_row(r) : r;
    int t   = threadIdx.x;
    float v    = in[(size_t)src * Cc + t];
    float mean = block_sum_256(v, sbuf) * (1.0f / Cc);
    float d    = v - mean;
    float var  = block_sum_256(d * d, sbuf) * (1.0f / Cc);
    float rstd = rsqrtf(var + 1e-6f);
    out[(size_t)r * Cc + t] = (_Float16)(d * rstd * g[t] + b[t]);
}

// ---------------------------------------------------------------------------
// Generic WMMA GEMM: C(M,N) = A(M,K) * Bt(N,K)^T + bias, epilogue variants.
// Block tile 128x64, 8 waves, each wave 2x2 16x16 subtiles, K step 32.
// Tiles staged with gfx1250 async copies (global_load_async_to_lds_b128,
// ASYNCcnt) into double-buffered LDS: stage tile k+1 while WMMAs consume k.
// ---------------------------------------------------------------------------
#define EPI_QKV  0
#define EPI_PROJ 1
#define EPI_GELU 2
#define EPI_ADD  3

struct GemmP {
    const _Float16* A;
    const _Float16* Bt;
    const float*    bias;
    int M, N, K, epi;
    float*          outf;   // PROJ / ADD
    const float*    resid;  // PROJ residual (x input)
    _Float16*       oq;     // QKV scatter
    _Float16*       ok;
    _Float16*       ov;
    _Float16*       outh;   // GELU f16 out
};

__global__ __launch_bounds__(256) void gemm_wmma(GemmP p) {
    __shared__ _Float16 aS[2][128 * 40];
    __shared__ _Float16 bS[2][64 * 40];
    const int m0   = blockIdx.x * 128;
    const int n0   = blockIdx.y * 64;
    const int tid  = threadIdx.x;
    const int wave = tid >> 5, lane = tid & 31;
    const int wm   = wave & 3, wn = wave >> 2;
    const int l16  = lane & 15;
    const int khalf = (lane < 16) ? 0 : 8;
    const int msub  = (lane >> 4) << 3;

    // staging geometry: per thread, two 8-half segments of A and one of B
    const int arow = tid >> 2;                 // 0..63  (+64 for second seg)
    const int aseg = (tid & 3) << 3;           // 0,8,16,24

    float8 acc[2][2];
#pragma unroll
    for (int a = 0; a < 2; ++a)
#pragma unroll
        for (int bb = 0; bb < 2; ++bb)
#pragma unroll
            for (int j = 0; j < 8; ++j) acc[a][bb][j] = 0.0f;

    // async-stage one 32-wide K tile of A (128 rows) and B (64 rows) into buf
    auto stage = [&](int k0, int buf) {
        const _Float16* ga0 = p.A + (size_t)(m0 + arow) * p.K + k0 + aseg;
        const _Float16* ga1 = p.A + (size_t)(m0 + arow + 64) * p.K + k0 + aseg;
        const _Float16* gb  = p.Bt + (size_t)(n0 + arow) * p.K + k0 + aseg;
        unsigned la0 = lds_off(&aS[buf][arow * 40 + aseg]);
        unsigned la1 = lds_off(&aS[buf][(arow + 64) * 40 + aseg]);
        unsigned lb  = lds_off(&bS[buf][arow * 40 + aseg]);
        asm volatile(
            "global_load_async_to_lds_b128 %0, %3, off\n\t"
            "global_load_async_to_lds_b128 %1, %4, off\n\t"
            "global_load_async_to_lds_b128 %2, %5, off"
            :: "v"(la0), "v"(la1), "v"(lb), "v"(ga0), "v"(ga1), "v"(gb)
            : "memory");
    };

    stage(0, 0);
    asm volatile("s_wait_asynccnt 0x0" ::: "memory");
    __syncthreads();

    const int nk = p.K >> 5;
    for (int kk = 0; kk < nk; ++kk) {
        const int cur = kk & 1;
        if (kk + 1 < nk) stage((kk + 1) << 5, cur ^ 1);

        const _Float16* aC = aS[cur];
        const _Float16* bC = bS[cur];
        half16 a[2], b[2];
#pragma unroll
        for (int im = 0; im < 2; ++im)
            a[im] = load_op(&aC[(wm * 32 + im * 16 + l16) * 40 + khalf]);
#pragma unroll
        for (int in = 0; in < 2; ++in)
            b[in] = load_op(&bC[(wn * 32 + in * 16 + l16) * 40 + khalf]);
#pragma unroll
        for (int im = 0; im < 2; ++im)
#pragma unroll
            for (int in = 0; in < 2; ++in)
                acc[im][in] = wmma_f16(a[im], b[in], acc[im][in]);

        asm volatile("s_wait_asynccnt 0x0" ::: "memory");
        __syncthreads();
    }

    // epilogue: lane holds (m = j + 8*(lane>=16), n = lane%16) per vgpr j
#pragma unroll
    for (int im = 0; im < 2; ++im) {
#pragma unroll
        for (int in = 0; in < 2; ++in) {
#pragma unroll
            for (int j = 0; j < 8; ++j) {
                int gm = m0 + wm * 32 + im * 16 + j + msub;
                int gn = n0 + wn * 32 + in * 16 + l16;
                float val = acc[im][in][j] + p.bias[gn];
                if (p.epi == EPI_QKV) {
                    int which = gn >> 8, cc = gn & 255;
                    int head = cc >> 5, d = cc & 31;
                    int w = gm / Nn, pos = gm - w * Nn;
                    size_t dst = (((size_t)w * NHh + head) * Nn + pos) * HDd + d;
                    _Float16* o = (which == 0) ? p.oq : (which == 1) ? p.ok : p.ov;
                    o[dst] = (_Float16)val;
                } else if (p.epi == EPI_PROJ) {
                    int dr = win_to_x_row(gm);
                    size_t idx = (size_t)dr * Cc + gn;
                    p.outf[idx] = p.resid[idx] + val;
                } else if (p.epi == EPI_GELU) {
                    float gv = 0.5f * val * (1.0f + erff(val * 0.70710678118654752f));
                    p.outh[(size_t)gm * p.N + gn] = (_Float16)gv;
                } else { // EPI_ADD
                    size_t idx = (size_t)gm * Cc + gn;
                    p.outf[idx] += val;
                }
            }
        }
    }
}

// ---------------------------------------------------------------------------
// Attention: one block per (window, head). q,k,v: (49,32) f16 each.
// S = q k^T (WMMA, padded to 64x64) -> fp32 softmax with diag/bias/mask
// -> P f16 -> O = P V (WMMA) -> o buffer (window-order rows, col head*32+d).
// ---------------------------------------------------------------------------
__global__ __launch_bounds__(128) void attn_kernel(const _Float16* __restrict__ qb,
                                                   const _Float16* __restrict__ kb,
                                                   const _Float16* __restrict__ vb,
                                                   const float* __restrict__ amask,
                                                   const float* __restrict__ rel_table,
                                                   const int* __restrict__ rpi,
                                                   const float* __restrict__ temperature,
                                                   _Float16* __restrict__ obuf) {
    __shared__ _Float16 qS[64 * 40];
    __shared__ _Float16 kS[64 * 40];
    __shared__ _Float16 vtS[32 * 72];
    __shared__ float    sS[64 * 65];
    __shared__ _Float16 pS[64 * 72];

    const int wi   = blockIdx.x >> 3;
    const int head = blockIdx.x & 7;
    const int tid  = threadIdx.x;
    const int wave = tid >> 5, lane = tid & 31;
    const int l16  = lane & 15;
    const int khalf = (lane < 16) ? 0 : 8;
    const int msub  = (lane >> 4) << 3;

    const size_t base = ((size_t)wi * NHh + head) * Nn * HDd;
    const _Float16* q = qb + base;
    const _Float16* k = kb + base;
    const _Float16* v = vb + base;

    for (int i = tid; i < 64 * 40; i += 128) { qS[i] = (_Float16)0.f; kS[i] = (_Float16)0.f; }
    for (int i = tid; i < 32 * 72; i += 128) vtS[i] = (_Float16)0.f;
    for (int i = tid; i < 64 * 72; i += 128) pS[i] = (_Float16)0.f;
    __syncthreads();
    for (int i = tid; i < Nn * HDd; i += 128) {
        int r = i >> 5, c = i & 31;
        qS[r * 40 + c]  = q[i];
        kS[r * 40 + c]  = k[i];
        vtS[c * 72 + r] = v[i];   // V^T for the B operand of P*V
    }
    __syncthreads();

    // S = q k^T : 4 waves * 4 n-tiles, single K step (K=32)
    {
        half16 aq = load_op(&qS[(wave * 16 + l16) * 40 + khalf]);
#pragma unroll
        for (int nt = 0; nt < 4; ++nt) {
            half16 bk = load_op(&kS[(nt * 16 + l16) * 40 + khalf]);
            float8 s;
#pragma unroll
            for (int j = 0; j < 8; ++j) s[j] = 0.0f;
            s = wmma_f16(aq, bk, s);
#pragma unroll
            for (int j = 0; j < 8; ++j) {
                int m = wave * 16 + j + msub;
                int n = nt * 16 + l16;
                sS[m * 65 + n] = s[j];
            }
        }
    }
    __syncthreads();

    // softmax over keys (cols 0..48), query rows 0..48; one thread per row
    const float scale = expf(temperature[0]);
    if (tid < Nn) {
        const int row = tid;
        const int*   rpir = rpi + row * Nn;
        const float* mrow = amask + ((size_t)(wi & 63) * Nn + row) * Nn;
        float mx = -3.0e38f;
        for (int m = 0; m < Nn; ++m) {
            float l;
            if (m == row) l = -1.0e30f;   // LSA diagonal mask (exp -> 0)
            else l = sS[row * 65 + m] * scale
                   + rel_table[rpir[m] * NHh + head] + mrow[m];
            sS[row * 65 + m] = l;
            mx = fmaxf(mx, l);
        }
        float sum = 0.0f;
        for (int m = 0; m < Nn; ++m) {
            float e = expf(sS[row * 65 + m] - mx);
            sS[row * 65 + m] = e;
            sum += e;
        }
        float inv = 1.0f / sum;
        for (int m = 0; m < Nn; ++m)
            pS[row * 72 + m] = (_Float16)(sS[row * 65 + m] * inv);
    }
    __syncthreads();

    // O = P(64x64) * V(64x32): 4 waves * 2 n-tiles * 2 K steps
#pragma unroll
    for (int nt = 0; nt < 2; ++nt) {
        float8 oacc;
#pragma unroll
        for (int j = 0; j < 8; ++j) oacc[j] = 0.0f;
#pragma unroll
        for (int ks = 0; ks < 2; ++ks) {
            int kb2 = ks * 32 + khalf;
            half16 ap = load_op(&pS[(wave * 16 + l16) * 72 + kb2]);
            half16 bv = load_op(&vtS[(nt * 16 + l16) * 72 + kb2]);
            oacc = wmma_f16(ap, bv, oacc);
        }
#pragma unroll
        for (int j = 0; j < 8; ++j) {
            int m = wave * 16 + j + msub;
            int n = nt * 16 + l16;
            if (m < Nn)
                obuf[((size_t)wi * Nn + m) * Cc + head * HDd + n] = (_Float16)oacc[j];
        }
    }
}

// ---------------------------------------------------------------------------
// Orchestration
// ---------------------------------------------------------------------------
extern "C" void kernel_launch(void* const* d_in, const int* in_sizes, int n_in,
                              void* d_out, int out_size, void* d_ws, size_t ws_size,
                              hipStream_t stream) {
    const float* x         = (const float*)d_in[0];
    const float* attn_mask = (const float*)d_in[1];
    const float* n1g       = (const float*)d_in[2];
    const float* n1b       = (const float*)d_in[3];
    const float* qkv_w     = (const float*)d_in[4];
    const float* qkv_b     = (const float*)d_in[5];
    const float* temp      = (const float*)d_in[6];
    const float* rel_table = (const float*)d_in[7];
    const float* proj_w    = (const float*)d_in[8];
    const float* proj_b    = (const float*)d_in[9];
    const float* n2g       = (const float*)d_in[10];
    const float* n2b       = (const float*)d_in[11];
    const float* fc1_w     = (const float*)d_in[12];
    const float* fc1_b     = (const float*)d_in[13];
    const float* fc2_w     = (const float*)d_in[14];
    const float* fc2_b     = (const float*)d_in[15];
    const int*   rpi       = (const int*)d_in[16];
    float*       out       = (float*)d_out;

    char* ws = (char*)d_ws;
    const size_t HALFS_MC = (size_t)MR * Cc;            // f16 slab: MR x 256
    const size_t R0_BYTES = (size_t)MR * 1024 * 2;      // fc1 output / q,k,v region
    const size_t R1_OFF   = R0_BYTES;                   // zw / h1 (f16, MR x 256)
    const size_t R2_OFF   = R1_OFF + HALFS_MC * 2;      // o (f16, MR x 256)
    const size_t WT_OFF   = R2_OFF + HALFS_MC * 2;      // converted weights

    _Float16* R0     = (_Float16*)ws;
    _Float16* zw     = (_Float16*)(ws + R1_OFF);        // later reused as h1
    _Float16* obuf   = (_Float16*)(ws + R2_OFF);
    _Float16* qkvwt  = (_Float16*)(ws + WT_OFF);
    _Float16* projwt = qkvwt + (size_t)768 * 256;
    _Float16* fc1wt  = projwt + (size_t)256 * 256;
    _Float16* fc2wt  = fc1wt + (size_t)1024 * 256;

    _Float16* qb = R0;
    _Float16* kb = qb + HALFS_MC;
    _Float16* vb = kb + HALFS_MC;

    // 1) weight convert/transpose to f16
    wt_convert<<<(768 * 256 + 255) / 256, 256, 0, stream>>>(qkv_w, qkvwt, 256, 768);
    wt_convert<<<(256 * 256 + 255) / 256, 256, 0, stream>>>(proj_w, projwt, 256, 256);
    wt_convert<<<(1024 * 256 + 255) / 256, 256, 0, stream>>>(fc1_w, fc1wt, 256, 1024);
    wt_convert<<<(1024 * 256 + 255) / 256, 256, 0, stream>>>(fc2_w, fc2wt, 1024, 256);

    // 2) LN1 + cyclic shift + window partition -> zw (f16, window-order rows)
    ln_kernel<<<MR, 256, 0, stream>>>(x, n1g, n1b, zw, 1);

    // 3) QKV GEMM (MR x 768), scatter into q/k/v (win,head,pos,d)
    {
        GemmP p{};
        p.A = zw; p.Bt = qkvwt; p.bias = qkv_b;
        p.M = MR; p.N = 768; p.K = 256; p.epi = EPI_QKV;
        p.oq = qb; p.ok = kb; p.ov = vb;
        gemm_wmma<<<dim3(MR / 128, 768 / 64), 256, 0, stream>>>(p);
    }

    // 4) windowed attention -> obuf (f16, window-order rows x 256)
    attn_kernel<<<TW * NHh, 128, 0, stream>>>(qb, kb, vb, attn_mask, rel_table,
                                              rpi, temp, obuf);

    // 5) proj GEMM + window reverse + roll + residual -> d_out (fp32 x1)
    {
        GemmP p{};
        p.A = obuf; p.Bt = projwt; p.bias = proj_b;
        p.M = MR; p.N = 256; p.K = 256; p.epi = EPI_PROJ;
        p.outf = out; p.resid = x;
        gemm_wmma<<<dim3(MR / 128, 256 / 64), 256, 0, stream>>>(p);
    }

    // 6) LN2(d_out) -> h1 (reuse zw region)
    ln_kernel<<<MR, 256, 0, stream>>>(out, n2g, n2b, zw, 0);

    // 7) FC1 + exact GELU -> R0 (f16, MR x 1024)
    {
        GemmP p{};
        p.A = zw; p.Bt = fc1wt; p.bias = fc1_b;
        p.M = MR; p.N = 1024; p.K = 256; p.epi = EPI_GELU;
        p.outh = R0;
        gemm_wmma<<<dim3(MR / 128, 1024 / 64), 256, 0, stream>>>(p);
    }

    // 8) FC2, accumulate into d_out: out = x1 + mlp
    {
        GemmP p{};
        p.A = R0; p.Bt = fc2wt; p.bias = fc2_b;
        p.M = MR; p.N = 256; p.K = 1024; p.epi = EPI_ADD;
        p.outf = out;
        gemm_wmma<<<dim3(MR / 128, 256 / 64), 256, 0, stream>>>(p);
    }

    (void)in_sizes; (void)n_in; (void)out_size; (void)ws_size;
}
